// OIMUnsupervisedLossOri_32916629357083
// MI455X (gfx1250) — matrix operand order repaired and checked by reading, not compiled, not run on hardware
//
#include <hip/hip_runtime.h>
#include <hip/hip_bf16.h>

#define NFEAT   256
#define NPIDS   15000
#define NROIS   4096
#define NTILES  938              // ceil(15000 / 16)
#define NPADC   (NTILES * 16)    // 15008 padded columns
#define OIMS    30.0f
#define IGNIDX  5554

typedef __attribute__((ext_vector_type(16))) __bf16    v16bf;
typedef __attribute__((ext_vector_type(8)))  float     v8f;
typedef __attribute__((ext_vector_type(4)))  unsigned  u32x4;

union BV { u32x4 q[2]; v16bf v; };   // 32 bytes = 16 bf16 (one WMMA operand per lane)

__device__ __forceinline__ unsigned short f2bf(float x) {
  unsigned u = __float_as_uint(x);
  unsigned r = u + 0x7FFFu + ((u >> 16) & 1u);   // round-to-nearest-even
  return (unsigned short)(r >> 16);
}

// ---------------------------------------------------------------------------
// Kernel 1: convert inputs (f32 -> bf16, PRE-SCALED by the OIM scalar so the
// GEMM emits logits directly) and lut (f32 -> bf16, zero-padded to NPADC
// rows). ~19 MB read, trivially bandwidth bound.
// ---------------------------------------------------------------------------
__global__ void oim_convert_kernel(const float* __restrict__ inp,
                                   const float* __restrict__ lut,
                                   unsigned short* __restrict__ Abf,
                                   unsigned short* __restrict__ Bbf) {
  const long stride = (long)gridDim.x * blockDim.x;
  const long tid    = (long)blockIdx.x * blockDim.x + threadIdx.x;
  const long nA  = (long)NROIS * NFEAT;
  const long nBv = (long)NPIDS * NFEAT;
  const long nB  = (long)NPADC * NFEAT;
  for (long i = tid; i < nA; i += stride) Abf[i] = f2bf(inp[i] * OIMS);
  for (long i = tid; i < nB; i += stride) Bbf[i] = (i < nBv) ? f2bf(lut[i]) : (unsigned short)0;
}

// ---------------------------------------------------------------------------
// Kernel 2: fused GEMM (bf16 WMMA) + online softmax + target-logit capture.
// One block = one 16-row M tile; 8 waves split the 938 N tiles round-robin.
// Each wave keeps the full 16x256 A tile resident (8 x v16bf = 64 VGPRs).
//
// Per-lane online softmax: each lane owns exactly ONE column per tile
// (col = n0 + lane%16), so running (max, sum) per row slot j needs no
// cross-lane traffic in the hot loop; the associative merge runs once at end.
// sched_barriers force: [16-load clause] -> [8 back-to-back WMMAs with
// partial loadcnt waits] -> [epilogue VALU], instead of load/wait/wmma
// ping-pong with only 2 loads in flight.
// ---------------------------------------------------------------------------
__global__ void __launch_bounds__(256, 1)
oim_fused_kernel(const unsigned short* __restrict__ Abf,
                 const unsigned short* __restrict__ Bbf,
                 const int* __restrict__ roi_label,
                 const int* __restrict__ labels,
                 float* __restrict__ nll_out,
                 float* __restrict__ msk_out) {
  const int m0     = blockIdx.x * 16;
  const int lane   = threadIdx.x & 31;
  const int wave   = threadIdx.x >> 5;   // 0..7
  const int l16    = lane & 15;
  const int laneHi = lane >> 4;          // which 16-lane half

  __shared__ int   lab_s[16];
  __shared__ int   val_s[16];
  __shared__ float red_m[8][16];
  __shared__ float red_s[8][16];
  __shared__ float red_t[8][16];

  if (threadIdx.x < 16) {
    int tgt  = roi_label[m0 + threadIdx.x] - 1;
    int safe = tgt < 0 ? 0 : tgt;
    lab_s[threadIdx.x] = labels[safe];
    val_s[threadIdx.x] = (tgt >= 0) ? 1 : 0;
  }
  __syncthreads();

  // --- Load A tile: lane l16 = row; per k-block of 32 the lane holds
  // K = kb*32 + laneHi*8 + {0..7} and K = kb*32 + 16 + laneHi*8 + {0..7}
  // (two contiguous 16-byte chunks, per the ISA 16-bit A layout).
  v16bf a[8];
  {
    const unsigned short* arow = Abf + (long)(m0 + l16) * NFEAT + laneHi * 8;
#pragma unroll
    for (int kb = 0; kb < 8; ++kb) {
      BV u;
      u.q[0] = *(const u32x4*)(arow + kb * 32);
      u.q[1] = *(const u32x4*)(arow + kb * 32 + 16);
      a[kb] = u.v;
    }
  }

  // Row this lane's slot j maps to: row = j + laneHi*8 (C/D layout).
  int labj[8];
#pragma unroll
  for (int j = 0; j < 8; ++j) labj[j] = lab_s[j + laneHi * 8];

  float rmax[8], rsum[8], tmax[8];
#pragma unroll
  for (int j = 0; j < 8; ++j) {
    rmax[j] = -__builtin_inff();
    rsum[j] = 0.f;
    tmax[j] = -__builtin_inff();
  }

  for (int t = wave; t < NTILES; t += 8) {
    const int n0 = t * 16;
    // B operand: lane l16 = output column; lane holds K = kb*32 + laneHi*8 + {0..7}
    // and K = kb*32 + 16 + laneHi*8 + {0..7} of that lut row (matches A).
    const unsigned short* bcol = Bbf + (long)(n0 + l16) * NFEAT + laneHi * 8;
    if (t + 8 < NTILES)
      __builtin_prefetch(Bbf + (long)(n0 + 128 + l16) * NFEAT + laneHi * 16, 0, 2);

    // Load ALL eight B k-blocks -> one 16-wide load clause.
    BV bu[8];
#pragma unroll
    for (int kb = 0; kb < 8; ++kb) {
      bu[kb].q[0] = *(const u32x4*)(bcol + kb * 32);
      bu[kb].q[1] = *(const u32x4*)(bcol + kb * 32 + 16);
    }

#if __has_builtin(__builtin_amdgcn_sched_barrier)
    __builtin_amdgcn_sched_barrier(0);   // keep all loads above the WMMAs
#endif

    v8f c = {0.f, 0.f, 0.f, 0.f, 0.f, 0.f, 0.f, 0.f};
#pragma unroll
    for (int kb = 0; kb < 8; ++kb)
      c = __builtin_amdgcn_wmma_f32_16x16x32_bf16(false, a[kb], false, bu[kb].v,
                                                  (short)0, c, false, false);

#if __has_builtin(__builtin_amdgcn_sched_barrier)
    __builtin_amdgcn_sched_barrier(0);   // keep epilogue below the WMMAs
#endif

    const int  col    = n0 + l16;
    const bool cvalid = (col < NPIDS);
#pragma unroll
    for (int j = 0; j < 8; ++j) {
      float v  = cvalid ? c[j] : -__builtin_inff();   // A pre-scaled by 30
      // Per-lane online softmax over the columns this lane owns.
      // (exp(-inf - -inf) cannot occur: padded cols only appear in the last
      // tile of wave 1, by which point rmax is finite.)
      float nm = fmaxf(rmax[j], v);
      rsum[j]  = rsum[j] * __expf(rmax[j] - nm) + __expf(v - nm);
      rmax[j]  = nm;
      // Target-logit capture: at most one owned column matches this row's label.
      tmax[j]  = (col == labj[j]) ? v : tmax[j];
    }
  }

  // --- One-time cross-lane merge (within each 16-lane half) of the
  // associative online-softmax state + target-logit max.
#pragma unroll
  for (int j = 0; j < 8; ++j) {
#pragma unroll
    for (int m = 1; m < 16; m <<= 1) {
      float om = __shfl_xor(rmax[j], m, 32);
      float os = __shfl_xor(rsum[j], m, 32);
      float nm = fmaxf(rmax[j], om);
      rsum[j]  = rsum[j] * __expf(rmax[j] - nm) + os * __expf(om - nm);
      rmax[j]  = nm;
      tmax[j]  = fmaxf(tmax[j], __shfl_xor(tmax[j], m, 32));
    }
  }

  // Publish per-wave partials (all lanes of a half now hold identical values).
  if (l16 == 0) {
#pragma unroll
    for (int j = 0; j < 8; ++j) {
      int r = j + laneHi * 8;
      red_m[wave][r] = rmax[j];
      red_s[wave][r] = rsum[j];
      red_t[wave][r] = tmax[j];
    }
  }
  __syncthreads();

  // Merge 8 wave partials per row, emit masked NLL.
  if (threadIdx.x < 16) {
    const int r = threadIdx.x;
    float m = -__builtin_inff(), s = 0.f, tt = -__builtin_inff();
    for (int w = 0; w < 8; ++w) {
      float mw = red_m[w][r], sw = red_s[w][r];
      float nm = fmaxf(m, mw);
      s = s * __expf(m - nm) + sw * __expf(mw - nm);
      m = nm;
      tt = fmaxf(tt, red_t[w][r]);
    }
    float nll = (m + __logf(s)) - tt;           // -(log_softmax at label)
    int msk = (val_s[r] != 0) && (lab_s[r] != IGNIDX);
    nll_out[m0 + r] = msk ? nll : 0.0f;
    msk_out[m0 + r] = (float)msk;
  }
}

// ---------------------------------------------------------------------------
// Kernel 3: reduce 4096 (nll, mask) pairs -> scalar loss.
// ---------------------------------------------------------------------------
__global__ void oim_finalize_kernel(const float* __restrict__ nll,
                                    const float* __restrict__ msk,
                                    float* __restrict__ out) {
  __shared__ float ss[256];
  __shared__ float sc[256];
  float s = 0.f, c = 0.f;
  for (int i = threadIdx.x; i < NROIS; i += 256) { s += nll[i]; c += msk[i]; }
  ss[threadIdx.x] = s;
  sc[threadIdx.x] = c;
  __syncthreads();
  for (int off = 128; off > 0; off >>= 1) {
    if (threadIdx.x < off) {
      ss[threadIdx.x] += ss[threadIdx.x + off];
      sc[threadIdx.x] += sc[threadIdx.x + off];
    }
    __syncthreads();
  }
  if (threadIdx.x == 0) out[0] = ss[0] / fmaxf(sc[0], 1.0f);
}

extern "C" void kernel_launch(void* const* d_in, const int* in_sizes, int n_in,
                              void* d_out, int out_size, void* d_ws, size_t ws_size,
                              hipStream_t stream) {
  const float* inputs    = (const float*)d_in[0];   // [4096, 256] f32
  const int*   roi_label = (const int*)d_in[1];     // [4096] i32
  const int*   labels    = (const int*)d_in[2];     // [15000] i32
  const float* lut       = (const float*)d_in[3];   // [15000, 256] f32
  float*       out       = (float*)d_out;           // scalar loss

  unsigned short* Abf = (unsigned short*)d_ws;                     // 2 MB
  unsigned short* Bbf = Abf + (long)NROIS * NFEAT;                 // ~7.7 MB
  float*          nll = (float*)(Bbf + (long)NPADC * NFEAT);       // 16 KB
  float*          msk = nll + NROIS;                               // 16 KB

  oim_convert_kernel<<<1024, 256, 0, stream>>>(inputs, lut, Abf, Bbf);
  oim_fused_kernel<<<NROIS / 16, 256, 0, stream>>>(Abf, Bbf, roi_label, labels, nll, msk);
  oim_finalize_kernel<<<1, 256, 0, stream>>>(nll, msk, out);
}
